// attention_FCN_29686813949997
// MI455X (gfx1250) — matrix operand order repaired and checked by reading, not compiled, not run on hardware
//
#include <hip/hip_runtime.h>
#include <hip/hip_bf16.h>

// ---------------- CDNA5 WMMA types ----------------
typedef __attribute__((ext_vector_type(16))) __bf16 v16bf;
typedef __attribute__((ext_vector_type(8)))  float  v8f;

union Frag { uint4 u[2]; v16bf v; };   // 32 bytes: 16 bf16 per lane (A or B operand)

// float -> bf16 round-to-nearest-even
__device__ __forceinline__ unsigned short f2bf(float f) {
    unsigned int u = __float_as_uint(f);
    unsigned int r = (u + 0x7fffu + ((u >> 16) & 1u)) >> 16;
    return (unsigned short)r;
}
__device__ __forceinline__ float bf2f(unsigned short h) {
    return __uint_as_float(((unsigned int)h) << 16);
}
__device__ __forceinline__ float bf2f_lo(unsigned int w) {
    return __uint_as_float(w << 16);
}
__device__ __forceinline__ float bf2f_hi(unsigned int w) {
    return __uint_as_float(w & 0xffff0000u);
}
// packed bf16x2 multiply (via f32); compiler lowers to v_pk_mul_f32 + pack
__device__ __forceinline__ unsigned int mul_bf16x2(unsigned int a, unsigned int b) {
    float a0 = bf2f_lo(a), a1 = bf2f_hi(a);
    float b0 = bf2f_lo(b), b1 = bf2f_hi(b);
    unsigned int r0 = __float_as_uint(a0 * b0) >> 16;
    unsigned int r1 = __float_as_uint(a1 * b1) & 0xffff0000u;
    return r1 | r0;
}
__device__ __forceinline__ float sigmoidf(float x) { return 1.f / (1.f + __expf(-x)); }

#define SF_STRIDE 136   // bf16 units per facts row in LDS (128 + pad, keeps 16B align)

// =====================================================================
// Kernel 1: q = PReLU(query@W1 + b1); term1 = q@(W2a+W2c) + b2  (s-independent fold)
// =====================================================================
__global__ __launch_bounds__(128) void qprep_kernel(
    const float* __restrict__ query, const float* __restrict__ W1,
    const float* __restrict__ b1,    const float* __restrict__ alpha,
    const float* __restrict__ W2,    const float* __restrict__ b2,
    unsigned short* __restrict__ qbf, float* __restrict__ term1)
{
    __shared__ float sq[128];
    __shared__ float sp[128];
    int b = blockIdx.x, j = threadIdx.x;
    sq[j] = query[b * 128 + j];
    __syncthreads();
    float acc = b1[j];
#pragma unroll 4
    for (int k = 0; k < 128; ++k) acc = fmaf(sq[k], W1[k * 128 + j], acc);
    float a  = alpha[j];
    float qv = acc >= 0.f ? acc : a * acc;          // PReLU
    qbf[b * 128 + j] = f2bf(qv);
    sp[j] = qv;
    __syncthreads();
    if (j < 80) {
        float t = b2[j];
#pragma unroll 4
        for (int k = 0; k < 128; ++k)
            t = fmaf(sp[k], W2[k * 80 + j] + W2[(256 + k) * 80 + j], t);
        term1[b * 80 + j] = t;
    }
}

// =====================================================================
// Kernel 2: pack folded weights into WMMA-B fragment layout (bf16)
//   wsW[0      .. 10240) : W2bc = W2b - W2c   (5 n-tiles x 4 k-steps x 512)
//   wsW[10240  .. 20480) : W2d               (5 x 4 x 512)
//   wsW[20480  .. 25088) : W3 padded 96x48   (3 n-tiles x 3 k-steps x 512)
// B fragment convention: element idx = lane*16 + e ; n_local = lane%16, k_local = (lane/16)*8 + (e&7) + (e>>3)*16
// =====================================================================
__global__ __launch_bounds__(256) void pack_weights_kernel(
    const float* __restrict__ W2, const float* __restrict__ W3,
    unsigned short* __restrict__ wsW)
{
    int tid = blockIdx.x * blockDim.x + threadIdx.x;
    int nthr = gridDim.x * blockDim.x;
    for (int idx = tid; idx < 5 * 4 * 512; idx += nthr) {
        int frag = idx >> 9, within = idx & 511;
        int ntile = frag >> 2, kstep = frag & 3;
        int l = within >> 4, e = within & 15;
        int k = kstep * 32 + (l >> 4) * 8 + (e & 7) + (e >> 3) * 16, n = ntile * 16 + (l & 15);
        float bc = W2[(128 + k) * 80 + n] - W2[(256 + k) * 80 + n];
        float dd = W2[(384 + k) * 80 + n];
        wsW[idx]          = f2bf(bc);
        wsW[10240 + idx]  = f2bf(dd);
    }
    for (int idx = tid; idx < 3 * 3 * 512; idx += nthr) {
        int frag = idx >> 9, within = idx & 511;
        int n2 = frag / 3, k2 = frag % 3;
        int l = within >> 4, e = within & 15;
        int k = k2 * 32 + (l >> 4) * 8 + (e & 7) + (e >> 3) * 16, n = n2 * 16 + (l & 15);
        float v = (k < 80 && n < 40) ? W3[k * 40 + n] : 0.f;
        wsW[20480 + idx] = f2bf(v);
    }
}

// =====================================================================
// Main fused kernel: one block per batch row b.
//  facts[b] -> bf16 in LDS (kept resident; read from HBM exactly once)
//  h1 = sigmoid(f@W2bc + (q*f)@W2d + term1)      [WMMA bf16, K=256]
//  h2 = sigmoid(h1@W3 + b3)                      [WMMA bf16, K=96 padded]
//  scores = h2@W4 + b4 ; masked softmax over S ; out = p @ facts
// =====================================================================
__global__ __launch_bounds__(256) void attn_main_kernel(
    const float* __restrict__ facts, const int* __restrict__ mask,
    const unsigned short* __restrict__ qbf, const float* __restrict__ term1,
    const unsigned short* __restrict__ wsW,
    const float* __restrict__ b3, const float* __restrict__ W4,
    const float* __restrict__ b4, float* __restrict__ out)
{
    extern __shared__ char smem[];
    unsigned short* sF      = (unsigned short*)(smem);            // 512*136*2 = 139264 B
    unsigned short* sW      = (unsigned short*)(smem + 139264);   // 50176 B (weight frags)
    unsigned short* sH1     = (unsigned short*)(smem + 189440);   // 8 waves * 16*96*2 = 24576 B
    float*          sPart   = (float*)(smem + 189440);            // reused: 16*128 f32 = 8192 B
    float*          sScores = (float*)(smem + 214016);            // 512 f32
    float*          sRed    = (float*)(smem + 216064);            // 16 f32

    const int b    = blockIdx.x;
    const int tid  = threadIdx.x;
    const int lane = tid & 31, wid = tid >> 5;
    const int lane16 = lane & 15, half = lane >> 4;

    // ---- stage weights fragments into LDS (6272 uint4 = 50176 u16)
    {
        const uint4* src = (const uint4*)wsW;
        uint4*       dst = (uint4*)sW;
        for (int i = tid; i < 6272; i += 256) dst[i] = src[i];
    }
    // ---- stream facts[b] (f32) -> bf16 LDS: 8 floats -> one ds_store_b128
    {
        const float4* f4 = (const float4*)(facts + (size_t)b * 512 * 128);
        for (int i = tid; i < 512 * 16; i += 256) {
            int row = i >> 4, c8 = i & 15;
            float4 v0 = f4[i * 2];
            float4 v1 = f4[i * 2 + 1];
            uint4 pk;
            pk.x = ((unsigned int)f2bf(v0.y) << 16) | f2bf(v0.x);
            pk.y = ((unsigned int)f2bf(v0.w) << 16) | f2bf(v0.z);
            pk.z = ((unsigned int)f2bf(v1.y) << 16) | f2bf(v1.x);
            pk.w = ((unsigned int)f2bf(v1.w) << 16) | f2bf(v1.z);
            *(uint4*)(sF + row * SF_STRIDE + c8 * 8) = pk;
        }
        // speculative prefetch of next block's facts into GL2
        __builtin_prefetch(facts + (size_t)(b + 1) * 512 * 128, 0, 1);
    }
    // ---- zero h1 staging (covers K-padding 80..95 once)
    {
        uint4 z = make_uint4(0, 0, 0, 0);
        uint4* h = (uint4*)sH1;
        for (int i = tid; i < 1536; i += 256) h[i] = z;
    }
    __syncthreads();

    // ---- per-wave q fragments (A-layout gather, held in regs across tiles)
    Frag qf[4];
    {
        const unsigned short* qrow = qbf + b * 128;
#pragma unroll
        for (int ks = 0; ks < 4; ++ks) {
            int k0 = ks * 32 + half * 8;
            qf[ks].u[0] = *(const uint4*)(qrow + k0);
            qf[ks].u[1] = *(const uint4*)(qrow + k0 + 16);
        }
    }
    const float b4v = b4[0];
    float w4r[3], b3r[3];
#pragma unroll
    for (int n2 = 0; n2 < 3; ++n2) {
        int n = n2 * 16 + lane16;
        w4r[n2] = (n < 40) ? W4[n] : 0.f;
        b3r[n2] = (n < 40) ? b3[n] : 0.f;
    }
    unsigned short* myH1 = sH1 + wid * 16 * 96;

    // ---- 4 s-tiles of 16 rows per wave (8 waves cover S=512)
    for (int t = 0; t < 4; ++t) {
        const int tile = wid * 4 + t;
        const int rowbase = tile * 16;
        v8f zero = {};
        v8f acc[5];
#pragma unroll
        for (int n = 0; n < 5; ++n) acc[n] = zero;

#pragma unroll
        for (int ks = 0; ks < 4; ++ks) {
            Frag af, aqf;
            const unsigned short* frow =
                sF + (rowbase + lane16) * SF_STRIDE + ks * 32 + half * 8;
            af.u[0] = *(const uint4*)(frow);
            af.u[1] = *(const uint4*)(frow + 16);
            {
                const unsigned int* fa = (const unsigned int*)&af;
                const unsigned int* qa = (const unsigned int*)&qf[ks];
                unsigned int*       oa = (unsigned int*)&aqf;
#pragma unroll
                for (int p = 0; p < 8; ++p) oa[p] = mul_bf16x2(fa[p], qa[p]);
            }
#pragma unroll
            for (int n = 0; n < 5; ++n) {
                Frag bbc, bd;
                const unsigned short* pbc = sW + (n * 4 + ks) * 512 + lane * 16;
                const unsigned short* pd  = sW + 10240 + (n * 4 + ks) * 512 + lane * 16;
                bbc.u[0] = *(const uint4*)pbc;  bbc.u[1] = *(const uint4*)(pbc + 8);
                bd.u[0]  = *(const uint4*)pd;   bd.u[1]  = *(const uint4*)(pd + 8);
                acc[n] = __builtin_amdgcn_wmma_f32_16x16x32_bf16(
                    false, af.v,  false, bbc.v, (short)0, acc[n], false, false);
                acc[n] = __builtin_amdgcn_wmma_f32_16x16x32_bf16(
                    false, aqf.v, false, bd.v,  (short)0, acc[n], false, false);
            }
        }
        // h1 = sigmoid(acc + term1)  -> bf16 staging in A-loadable row-major [16][96]
#pragma unroll
        for (int n = 0; n < 5; ++n) {
            float t1 = term1[b * 80 + n * 16 + lane16];
#pragma unroll
            for (int v = 0; v < 8; ++v) {
                float h = sigmoidf(acc[n][v] + t1);
                myH1[(v + 8 * half) * 96 + n * 16 + lane16] = f2bf(h);
            }
        }
        asm volatile("s_wait_dscnt 0" ::: "memory");   // wave-local LDS RAW fence

        // h2 = sigmoid(h1 @ W3 + b3)   (M=16, K=96 padded, N=48 padded)
        v8f acc2[3];
#pragma unroll
        for (int n = 0; n < 3; ++n) acc2[n] = zero;
#pragma unroll
        for (int k2 = 0; k2 < 3; ++k2) {
            Frag ah;
            const unsigned short* hrow = myH1 + lane16 * 96 + k2 * 32 + half * 8;
            ah.u[0] = *(const uint4*)hrow;
            ah.u[1] = *(const uint4*)(hrow + 16);
#pragma unroll
            for (int n2 = 0; n2 < 3; ++n2) {
                Frag bw;
                const unsigned short* pw = sW + 20480 + (n2 * 3 + k2) * 512 + lane * 16;
                bw.u[0] = *(const uint4*)pw;  bw.u[1] = *(const uint4*)(pw + 8);
                acc2[n2] = __builtin_amdgcn_wmma_f32_16x16x32_bf16(
                    false, ah.v, false, bw.v, (short)0, acc2[n2], false, false);
            }
        }
        // scores(m) = sum_n sigmoid(h2) * W4[n]  (cross-lane reduce over N)
        float pv[8];
#pragma unroll
        for (int v = 0; v < 8; ++v) pv[v] = 0.f;
#pragma unroll
        for (int n2 = 0; n2 < 3; ++n2) {
#pragma unroll
            for (int v = 0; v < 8; ++v) {
                float h2 = sigmoidf(acc2[n2][v] + b3r[n2]);
                pv[v] = fmaf(h2, w4r[n2], pv[v]);
            }
        }
#pragma unroll
        for (int v = 0; v < 8; ++v) {
#pragma unroll
            for (int off = 8; off >= 1; off >>= 1)
                pv[v] += __shfl_xor(pv[v], off, 32);
        }
        if (lane16 == 0) {
#pragma unroll
            for (int v = 0; v < 8; ++v)
                sScores[rowbase + v + 8 * half] = pv[v] + b4v;
        }
    }
    __syncthreads();

    // ---- masked softmax over S = 512 (2 entries per thread)
    const float PADV = -4294967295.f;
    float sc0 = sScores[tid], sc1 = sScores[tid + 256];
    if (mask[b * 512 + tid] != 1)       sc0 = PADV;
    if (mask[b * 512 + tid + 256] != 1) sc1 = PADV;
    float mx = fmaxf(sc0, sc1);
#pragma unroll
    for (int off = 16; off >= 1; off >>= 1) mx = fmaxf(mx, __shfl_xor(mx, off, 32));
    if (lane == 0) sRed[wid] = mx;
    __syncthreads();
    mx = sRed[0];
#pragma unroll
    for (int i = 1; i < 8; ++i) mx = fmaxf(mx, sRed[i]);
    float e0 = __expf(sc0 - mx), e1 = __expf(sc1 - mx);
    float sum = e0 + e1;
#pragma unroll
    for (int off = 16; off >= 1; off >>= 1) sum += __shfl_xor(sum, off, 32);
    if (lane == 0) sRed[8 + wid] = sum;
    __syncthreads();
    float tot = 0.f;
#pragma unroll
    for (int i = 0; i < 8; ++i) tot += sRed[8 + i];
    float inv = 1.f / tot;
    sScores[tid]       = e0 * inv;
    sScores[tid + 256] = e1 * inv;
    __syncthreads();

    // ---- out[b,f] = sum_s p[s] * facts_bf16[s,f]   (facts resident in LDS)
    // Vectorized: thread owns 8-column group (cg) x 32-row segment (seg);
    // each row is one ds_load_b128, then a 16-way LDS partial reduction.
    {
        const int cg  = tid & 15;    // columns cg*8 .. cg*8+7
        const int seg = tid >> 4;    // rows seg*32 .. seg*32+31
        float a[8];
#pragma unroll
        for (int j = 0; j < 8; ++j) a[j] = 0.f;
        const unsigned short* basec = sF + cg * 8;
        for (int s = seg * 32; s < seg * 32 + 32; ++s) {
            float p = sScores[s];
            uint4 w = *(const uint4*)(basec + s * SF_STRIDE);
            a[0] = fmaf(p, bf2f_lo(w.x), a[0]);
            a[1] = fmaf(p, bf2f_hi(w.x), a[1]);
            a[2] = fmaf(p, bf2f_lo(w.y), a[2]);
            a[3] = fmaf(p, bf2f_hi(w.y), a[3]);
            a[4] = fmaf(p, bf2f_lo(w.z), a[4]);
            a[5] = fmaf(p, bf2f_hi(w.z), a[5]);
            a[6] = fmaf(p, bf2f_lo(w.w), a[6]);
            a[7] = fmaf(p, bf2f_hi(w.w), a[7]);
        }
        float* myp = sPart + seg * 128 + cg * 8;
        *(float4*)(myp)     = make_float4(a[0], a[1], a[2], a[3]);
        *(float4*)(myp + 4) = make_float4(a[4], a[5], a[6], a[7]);
    }
    __syncthreads();
    if (tid < 128) {
        float o = 0.f;
#pragma unroll
        for (int g = 0; g < 16; ++g) o += sPart[g * 128 + tid];
        out[(size_t)b * 128 + tid] = o;
    }
}

// =====================================================================
extern "C" void kernel_launch(void* const* d_in, const int* in_sizes, int n_in,
                              void* d_out, int out_size, void* d_ws, size_t ws_size,
                              hipStream_t stream) {
    const float* query = (const float*)d_in[0];
    const float* facts = (const float*)d_in[1];
    const int*   mask  = (const int*)d_in[2];
    const float* W1    = (const float*)d_in[3];
    const float* b1    = (const float*)d_in[4];
    const float* alpha = (const float*)d_in[5];
    const float* W2    = (const float*)d_in[6];
    const float* b2    = (const float*)d_in[7];
    const float* W3    = (const float*)d_in[8];
    const float* b3    = (const float*)d_in[9];
    const float* W4    = (const float*)d_in[10];
    const float* b4    = (const float*)d_in[11];
    float* out = (float*)d_out;

    char* ws = (char*)d_ws;
    unsigned short* qbf   = (unsigned short*)ws;                       // 131072 B
    float*          term1 = (float*)(ws + 131072);                     // 163840 B
    unsigned short* wsW   = (unsigned short*)(ws + 131072 + 163840);   // 50176 B

    qprep_kernel<<<512, 128, 0, stream>>>(query, W1, b1, alpha, W2, b2, qbf, term1);
    pack_weights_kernel<<<40, 256, 0, stream>>>(W2, W3, wsW);

    size_t shmem = 216640;   // ~212 KB dynamic LDS (CDNA5 WGP allows 320 KB)
    attn_main_kernel<<<512, 256, shmem, stream>>>(facts, mask, qbf, term1, wsW,
                                                  b3, W4, b4, out);
}
